// MessageLayer_41875931136229
// MI455X (gfx1250) — matrix operand order repaired and checked by reading, not compiled, not run on hardware
//
#include <hip/hip_runtime.h>

typedef __attribute__((ext_vector_type(16))) _Float16     v16h;
typedef __attribute__((ext_vector_type(8)))  float        v8f;
typedef __attribute__((ext_vector_type(4)))  float        v4f;
typedef __attribute__((ext_vector_type(4)))  unsigned int v4u;
typedef __attribute__((ext_vector_type(4)))  int          v4i;

#define N_NODES   10000
#define N_EDGES   320000
#define HIDDEN    16
#define MSG       64
#define EDGE_FEAT 32
#define N_TILES   (N_EDGES / 16)     // 20000 tiles of 16 edges
#define N_PAIRS   (N_TILES / 2)      // 2 tiles per wave iteration
#define N_CHUNK   17                 // 16 K-chunks of W (K=512) + 1 bias chunk
#define NTCOL     4                  // 64 msg cols / 16
#define B_ELEMS   (N_CHUNK * NTCOL * 512)   // halves in packed B stream
#define B_BYTES   (B_ELEMS * 2)             // 69632 bytes of LDS

// Pointer types matching the async-to-LDS builtin's parameters:
// param0: int4 pointer in global (device) address space, param1: int4 in LDS.
typedef __attribute__((address_space(1))) v4i* gv4_t;
typedef __attribute__((address_space(3))) v4i* lv4_t;

// ---------------------------------------------------------------------------
// Kernel 0: zero the output accumulator table (harness poisons d_out).
// ---------------------------------------------------------------------------
__global__ void zero_kernel(float* __restrict__ p, int n) {
    int i = blockIdx.x * blockDim.x + threadIdx.x;
    if (i < n) p[i] = 0.0f;
}

// ---------------------------------------------------------------------------
// Kernel 1: repack W (fp32 [32, 64*16]) + bias b into the exact per-lane
// B-operand stream for v_wmma_f32_16x16x32_f16, as fp16, into d_ws.
// Stream layout: block = c*4 + n  (c = K-chunk, n = msg-col tile),
//   within block: lane (0..31) * 16 halves, half i holds
//   B[K = 16*(i>=8) + 8*(lane>=16) + i%8, m = n*16 + lane%16] of chunk c.
// Chunk c<16 covers kh = 32c..32c+31 with kh = k*16 + h, i.e. k in {2c,2c+1}.
// Chunk 16 is the bias: rows K<16 = b[m*16+K], rows K>=16 = 0.
// ---------------------------------------------------------------------------
__global__ void prepack_kernel(const float* __restrict__ W,
                               const float* __restrict__ b,
                               _Float16* __restrict__ wr) {
    int idx = blockIdx.x * blockDim.x + threadIdx.x;
    if (idx >= B_ELEMS) return;
    int i    = idx & 15;
    int lane = (idx >> 4) & 31;
    int blk  = idx >> 9;
    int c = blk >> 2, n = blk & 3;
    int Kl = ((i >> 3) << 4) + ((lane >> 4) << 3) + (i & 7);  // 0..31 within chunk
    int m  = (n << 4) + (lane & 15);          // message column 0..63
    float v;
    if (c < 16) {
        int kh = (c << 5) + Kl;
        int k = kh >> 4, h = kh & 15;
        v = W[k * (MSG * HIDDEN) + m * HIDDEN + h];
    } else {
        v = (Kl < HIDDEN) ? b[m * HIDDEN + Kl] : 0.0f;
    }
    wr[idx] = (_Float16)v;
}

// ---------------------------------------------------------------------------
// Kernel 2: main message GEMM. Each wave handles pairs of 16-edge tiles:
//   D[16 edges, 64 msgs] += Outer(ef, nh)[16, 512+16] x B[512+16, 64]
// A is built on the fly (2 fp32 muls + pack per 2 WMMA slots), B comes from
// LDS (staged via async global->LDS loads), accumulation in fp32, scatter via
// global_atomic_add_f32.
// ---------------------------------------------------------------------------
__global__ __launch_bounds__(256)
void msg_kernel(const float* __restrict__ ef,  const int* __restrict__ esrc,
                const int* __restrict__ etgt,  const float* __restrict__ hid,
                const _Float16* __restrict__ wr, float* __restrict__ out) {
    extern __shared__ char smem[];
    // 69.6 KB B-panel fill: global -> LDS, preferring CDNA5 async-to-LDS path
    // (tracked by ASYNCcnt, no VGPR round-trip).
    {
        v4i* g = (v4i*)wr;
        v4i* s = (v4i*)smem;
#if __has_builtin(__builtin_amdgcn_global_load_async_to_lds_b128)
        for (int i = threadIdx.x; i < B_BYTES / 16; i += blockDim.x) {
            __builtin_amdgcn_global_load_async_to_lds_b128(
                (gv4_t)(g + i), (lv4_t)(s + i), 0, 0);
        }
#if __has_builtin(__builtin_amdgcn_s_wait_asynccnt)
        __builtin_amdgcn_s_wait_asynccnt(0);
#else
        asm volatile("s_wait_asynccnt 0" ::: "memory");
#endif
#else
        for (int i = threadIdx.x; i < B_BYTES / 16; i += blockDim.x) s[i] = g[i];
#endif
    }
    __syncthreads();

    const int lane = threadIdx.x & 31;
    const int hi   = lane >> 4;        // which half-wave (K/M split)
    const int m16  = lane & 15;
    const int wid  = (blockIdx.x * blockDim.x + threadIdx.x) >> 5;
    const int nw   = (gridDim.x * blockDim.x) >> 5;

    for (int tp = wid; tp < N_PAIRS; tp += nw) {
        const int tile0 = tp * 2;
        if (tp + nw < N_PAIRS)   // prefetch next pair's edge features
            __builtin_prefetch(ef + (long)(tp + nw) * 2 * 16 * EDGE_FEAT, 0, 0);

        float nhf[2][8];
        float efr[2][32];
        v8f   acc[2][4] = {};
#pragma unroll
        for (int t = 0; t < 2; ++t) {
            const int ea  = (tile0 + t) * 16 + m16;   // this lane's A-edge
            const int src = esrc[ea];
            const float* hp = hid + (long)src * HIDDEN + hi * 8;
            *(v4f*)&nhf[t][0] = *(const v4f*)hp;
            *(v4f*)&nhf[t][4] = *(const v4f*)(hp + 4);
            const v4f* ep = (const v4f*)(ef + (long)ea * EDGE_FEAT);
#pragma unroll
            for (int q = 0; q < 8; ++q) *(v4f*)&efr[t][4 * q] = ep[q];
        }

        for (int c = 0; c < N_CHUNK; ++c) {
            // Build A (16x32 fp16 operand) for both tiles.
            // ISA layout: lanes<16 hold K 0-7 / 16-23, lanes>=16 hold K 8-15 / 24-31;
            // with kh = k*16+h this means: halves 0..7 = a0*nh[base+j],
            // halves 8..15 = a1*nh[base+j], base = hi*8 (already folded into nhf).
            v16h A[2];
#pragma unroll
            for (int t = 0; t < 2; ++t) {
                const float a0 = (c < 16) ? efr[t][2 * c]     : 1.0f;
                const float a1 = (c < 16) ? efr[t][2 * c + 1] : 0.0f;
#pragma unroll
                for (int j = 0; j < 8; ++j) {
                    A[t][j]     = (_Float16)(a0 * nhf[t][j]);
                    A[t][j + 8] = (_Float16)(a1 * nhf[t][j]);
                }
            }
#pragma unroll
            for (int n = 0; n < 4; ++n) {
                union { v16h h; v4u u[2]; } B;
                const char* bp = smem + (((c << 2) + n) << 10) + (lane << 5);
                B.u[0] = *(const v4u*)bp;
                B.u[1] = *(const v4u*)(bp + 16);
#pragma unroll
                for (int t = 0; t < 2; ++t)
                    acc[t][n] = __builtin_amdgcn_wmma_f32_16x16x32_f16(
                        false, A[t], false, B.h, (short)0, acc[t][n], false, false);
            }
        }

        // Scatter-add. D layout: lane -> N = lane%16; VGPR r -> M = r + hi*8.
#pragma unroll
        for (int t = 0; t < 2; ++t) {
            const int tb = (tile0 + t) * 16 + hi * 8;
#pragma unroll
            for (int r = 0; r < 8; ++r) {
                const long row = (long)etgt[tb + r] * MSG;
#pragma unroll
                for (int n = 0; n < 4; ++n) {
                    float* dst = out + row + n * 16 + m16;
                    __hip_atomic_fetch_add(dst, acc[t][n][r],
                                           __ATOMIC_RELAXED, __HIP_MEMORY_SCOPE_AGENT);
                }
            }
        }
    }
}

// ---------------------------------------------------------------------------
extern "C" void kernel_launch(void* const* d_in, const int* in_sizes, int n_in,
                              void* d_out, int out_size, void* d_ws, size_t ws_size,
                              hipStream_t stream) {
    // setup_inputs order:
    // 0 node_features (unused by reference output), 1 edge_features,
    // 2 edge_sources, 3 edge_targets, 4 hidden, 5 initial (unused), 6 W, 7 b
    const float* edge_features = (const float*)d_in[1];
    const int*   edge_sources  = (const int*)d_in[2];
    const int*   edge_targets  = (const int*)d_in[3];
    const float* hidden        = (const float*)d_in[4];
    const float* W             = (const float*)d_in[6];
    const float* b             = (const float*)d_in[7];
    float*     out = (float*)d_out;
    _Float16*  wr  = (_Float16*)d_ws;   // needs 69632 bytes

    zero_kernel<<<(out_size + 255) / 256, 256, 0, stream>>>(out, out_size);
    prepack_kernel<<<(B_ELEMS + 255) / 256, 256, 0, stream>>>(W, b, wr);
    msg_kernel<<<256, 256, B_BYTES, stream>>>(edge_features, edge_sources,
                                              edge_targets, hidden, wr, out);
}
